// AttentionBlock_9010841387434
// MI455X (gfx1250) — matrix operand order repaired and checked by reading, not compiled, not run on hardware
//
#include <hip/hip_runtime.h>

#define CH 512
#define HW 1024
#define NH 4
#define HD 128

typedef __attribute__((ext_vector_type(16))) _Float16 v16h;
typedef __attribute__((ext_vector_type(8)))  _Float16 v8h;
typedef __attribute__((ext_vector_type(8)))  float    v8f;

// ---------------------------------------------------------------------------
// WMMA fragment helpers (layouts per CDNA5 ISA 7.12.2, wave32)
// ---------------------------------------------------------------------------
__device__ __forceinline__ v8f vzero8() {
  v8f z;
#pragma unroll
  for (int i = 0; i < 8; ++i) z[i] = 0.0f;
  return z;
}

// A-matrix 16x32 f16 tile from LDS, row-major, row stride `ld` (mult of 8).
__device__ __forceinline__ v16h frag_a(const _Float16* base, int ld) {
  const int lane = threadIdx.x & 31;
  const int m = lane & 15;
  const int sel = lane >> 4;
  const _Float16* p = base + m * ld + sel * 8;
  v8h lo = *(const v8h*)(p);
  v8h hi = *(const v8h*)(p + 16);
  return __builtin_shufflevector(lo, hi, 0, 1, 2, 3, 4, 5, 6, 7, 8, 9, 10, 11, 12, 13, 14, 15);
}

// B-matrix 32x16 f16 tile stored transposed in LDS as [N=16][K=32] rows, stride `ld`.
__device__ __forceinline__ v16h frag_b(const _Float16* base, int ld) {
  const int lane = threadIdx.x & 31;
  const int n = lane & 15;
  const int sel = lane >> 4;
  const _Float16* p = base + n * ld + sel * 8;
  v8h lo = *(const v8h*)(p);
  v8h hi = *(const v8h*)(p + 16);
  return __builtin_shufflevector(lo, hi, 0, 1, 2, 3, 4, 5, 6, 7, 8, 9, 10, 11, 12, 13, 14, 15);
}

__device__ __forceinline__ v8f wmma16(v16h a, v16h b, v8f c) {
  return __builtin_amdgcn_wmma_f32_16x16x32_f16(false, a, false, b, (short)0, c, false, false);
}

// ---------------------------------------------------------------------------
// Async global->LDS copy (CDNA5 GLOBAL_LOAD_ASYNC_TO_LDS_B128, ASYNCcnt).
// The instruction offset applies to BOTH the LDS and global address, so one
// (lds, vaddr) register pair covers a 128-byte contiguous run per lane.
// ---------------------------------------------------------------------------
#define ASYNC_LD(ldsv, vav, sb, OFF)                                          \
  asm volatile("global_load_async_to_lds_b128 %0, %1, %2 offset:" #OFF        \
               :: "v"(ldsv), "v"(vav), "s"(sb) : "memory")

__device__ __forceinline__ void async_copy_128B(unsigned ldsoff, unsigned va,
                                                const _Float16* sb) {
  ASYNC_LD(ldsoff, va, sb, 0);
  ASYNC_LD(ldsoff, va, sb, 16);
  ASYNC_LD(ldsoff, va, sb, 32);
  ASYNC_LD(ldsoff, va, sb, 48);
  ASYNC_LD(ldsoff, va, sb, 64);
  ASYNC_LD(ldsoff, va, sb, 80);
  ASYNC_LD(ldsoff, va, sb, 96);
  ASYNC_LD(ldsoff, va, sb, 112);
}

__device__ __forceinline__ void wait_async0() {
  asm volatile("s_wait_asynccnt 0x0" ::: "memory");
}
__device__ __forceinline__ void wait_ds0() {
  asm volatile("s_wait_dscnt 0x0" ::: "memory");
}

// ---------------------------------------------------------------------------
// Kernel 0: f32 -> f16 weight conversion
// ---------------------------------------------------------------------------
__global__ __launch_bounds__(256) void k_f32tof16(const float* __restrict__ in,
                                                  _Float16* __restrict__ out, int n) {
  int i = blockIdx.x * 256 + threadIdx.x;
  if (i < n) out[i] = (_Float16)in[i];
}

// ---------------------------------------------------------------------------
// Kernel 1: GroupNorm -> xnT [b][hw=1024][c=512] f16 (transposed for GEMM B)
// ---------------------------------------------------------------------------
__global__ __launch_bounds__(256) void k_groupnorm(const float* __restrict__ x,
                                                   const float* __restrict__ gw,
                                                   const float* __restrict__ gb,
                                                   _Float16* __restrict__ xnT) {
  const int b = blockIdx.x >> 5;
  const int g = blockIdx.x & 31;
  const float* xp = x + ((size_t)b * CH + g * 16) * HW;

  float s = 0.f, s2 = 0.f;
  for (int i = threadIdx.x; i < 16 * HW; i += 256) {
    float v = xp[i];
    s += v;
    s2 += v * v;
  }
  __shared__ float red0[256];
  __shared__ float red1[256];
  red0[threadIdx.x] = s;
  red1[threadIdx.x] = s2;
  __syncthreads();
  for (int off = 128; off > 0; off >>= 1) {
    if (threadIdx.x < off) {
      red0[threadIdx.x] += red0[threadIdx.x + off];
      red1[threadIdx.x] += red1[threadIdx.x + off];
    }
    __syncthreads();
  }
  const float inv_n = 1.0f / (16.0f * HW);
  const float mean = red0[0] * inv_n;
  const float var = red1[0] * inv_n - mean * mean;
  const float rstd = __frsqrt_rn(var + 1e-5f);

  __shared__ float sc[16];
  __shared__ float sh[16];
  if (threadIdx.x < 16) {
    float gma = gw[g * 16 + threadIdx.x];
    float bta = gb[g * 16 + threadIdx.x];
    sc[threadIdx.x] = gma * rstd;
    sh[threadIdx.x] = bta - mean * rstd * gma;
  }
  __syncthreads();

#pragma unroll
  for (int j = 0; j < 4; ++j) {
    const int hw = threadIdx.x + j * 256;
    alignas(16) _Float16 outv[16];
#pragma unroll
    for (int ch = 0; ch < 16; ++ch) {
      float v = xp[(size_t)ch * HW + hw];
      outv[ch] = (_Float16)(v * sc[ch] + sh[ch]);
    }
    _Float16* dst = xnT + ((size_t)b * HW + hw) * CH + g * 16;
    ((v8h*)dst)[0] = ((const v8h*)outv)[0];
    ((v8h*)dst)[1] = ((const v8h*)outv)[1];
  }
}

// ---------------------------------------------------------------------------
// Kernel 2: QKV GEMM  (per batch: W[1536x512] x Xn[512x1024])
// ---------------------------------------------------------------------------
#define LDT 40  // LDS stride for 32-wide K tiles (padded, mult of 8)

__global__ __launch_bounds__(256) void k_gemm_qkv(const _Float16* __restrict__ W,
                                                  const _Float16* __restrict__ X,
                                                  const float* __restrict__ bias,
                                                  _Float16* __restrict__ qT,
                                                  _Float16* __restrict__ kT,
                                                  _Float16* __restrict__ V) {
  __shared__ _Float16 As[128 * LDT];
  __shared__ _Float16 Bs[128 * LDT];
  const int b = blockIdx.z;
  const int M0 = blockIdx.x * 128;
  const int N0 = blockIdx.y * 128;
  const int wv = threadIdx.x >> 5;
  const int lane = threadIdx.x & 31;
  const _Float16* Xb = X + (size_t)b * HW * CH;

  v8f acc[8];
#pragma unroll
  for (int i = 0; i < 8; ++i) acc[i] = vzero8();

  const int lrow = threadIdx.x >> 1;
  const int lcol = (threadIdx.x & 1) * 16;

  for (int k0 = 0; k0 < CH; k0 += 32) {
    __syncthreads();
    {
      const v8h* srcA = (const v8h*)(W + (size_t)(M0 + lrow) * CH + k0 + lcol);
      v8h* dstA = (v8h*)(As + lrow * LDT + lcol);
      dstA[0] = srcA[0];
      dstA[1] = srcA[1];
      const v8h* srcB = (const v8h*)(Xb + (size_t)(N0 + lrow) * CH + k0 + lcol);
      v8h* dstB = (v8h*)(Bs + lrow * LDT + lcol);
      dstB[0] = srcB[0];
      dstB[1] = srcB[1];
    }
    __syncthreads();
    v16h a = frag_a(As + (wv * 16) * LDT, LDT);
#pragma unroll
    for (int nt = 0; nt < 8; ++nt) {
      v16h bf = frag_b(Bs + (nt * 16) * LDT, LDT);
      acc[nt] = wmma16(a, bf, acc[nt]);
    }
  }

  const int region = M0 >> 9;        // 0=q 1=k 2=v
  const int head = (M0 & 511) >> 7;
  const int nl = lane & 15;
  const int moff = (lane >> 4) * 8;  // D-frag: lane<16 -> M=r, lane>=16 -> M=r+8

  if (region < 2) {
    _Float16* dstb = (region == 0 ? qT : kT) + (size_t)(b * NH + head) * HW * HD;
#pragma unroll
    for (int nt = 0; nt < 8; ++nt) {
      const int hw = N0 + nt * 16 + nl;
      alignas(16) _Float16 tmp[8];
#pragma unroll
      for (int i = 0; i < 8; ++i)
        tmp[i] = (_Float16)(acc[nt][i] + bias[M0 + wv * 16 + moff + i]);
      *(v8h*)(dstb + (size_t)hw * HD + wv * 16 + moff) = *(const v8h*)tmp;
    }
  } else {
    _Float16* dstb = V + (size_t)(b * NH + head) * HD * HW;
#pragma unroll
    for (int nt = 0; nt < 8; ++nt) {
      const int hw = N0 + nt * 16 + nl;
#pragma unroll
      for (int r = 0; r < 8; ++r) {
        const int d = wv * 16 + r + moff;
        dstb[(size_t)d * HW + hw] = (_Float16)(acc[nt][r] + bias[M0 + d]);
      }
    }
  }
}

// ---------------------------------------------------------------------------
// Kernel 3: fused flash attention per (bh, 128-row m-block), grid (8, 64).
// Async TDM-path pipelining: V DMA overlaps S-gemm+softmax, next-K DMA
// overlaps O-gemm. S never leaves LDS. Writes aoT [b][hw][c] f16.
// ---------------------------------------------------------------------------
#define FPAD 136  // f16 LDS stride for 128-wide tiles (mult of 8)

__global__ __launch_bounds__(256) void k_attn(const _Float16* __restrict__ qT,
                                              const _Float16* __restrict__ kT,
                                              const _Float16* __restrict__ V,
                                              _Float16* __restrict__ aoT) {
  __shared__ _Float16 Qs[128 * FPAD];
  __shared__ _Float16 Ks[128 * FPAD];
  __shared__ _Float16 Vs[128 * FPAD];
  __shared__ _Float16 Ssh[8][16 * FPAD];  // per-wave S/P tile (f16)
  __shared__ float rowM[8][16];
  __shared__ float rowL[8][16];
  __shared__ float rowC[8][16];
  __shared__ float pmax[8][32];
  __shared__ float psum[8][32];

  const int bh = blockIdx.y;
  const int m0 = blockIdx.x * 128;
  const int wv = threadIdx.x >> 5;
  const int lane = threadIdx.x & 31;
  const _Float16* Qg = qT + (size_t)bh * HW * HD;
  const _Float16* Kg = kT + (size_t)bh * HW * HD;
  const _Float16* Vg = V + (size_t)bh * HD * HW;

  const int lrow = threadIdx.x >> 1;        // 0..127
  const int lcol = (threadIdx.x & 1) * 64;  // 0 or 64

  // LDS byte offsets for this thread's 128B run (generic ptr low 32 bits)
  const unsigned ldsQ = (unsigned)(size_t)&Qs[lrow * FPAD + lcol];
  const unsigned ldsK = (unsigned)(size_t)&Ks[lrow * FPAD + lcol];
  const unsigned ldsV = (unsigned)(size_t)&Vs[lrow * FPAD + lcol];

  // kick off Q block + first K tile
  async_copy_128B(ldsQ, (unsigned)(((m0 + lrow) * HD + lcol) * 2), Qg);
  async_copy_128B(ldsK, (unsigned)((lrow * HD + lcol) * 2), Kg);

  if (lane < 16) {
    rowM[wv][lane] = -1e30f;
    rowL[wv][lane] = 0.0f;
  }

  v8f o[8];
#pragma unroll
  for (int i = 0; i < 8; ++i) o[i] = vzero8();

  const int nl = lane & 15;
  const int ro = (lane >> 4) * 8;
  const float scale = 0.088388347648318447f;  // 1/sqrt(128)

  for (int n0 = 0; n0 < HW; n0 += 128) {
    wait_async0();
    __syncthreads();  // K tile (and Q) visible to all waves; Vs free

    // prefetch V tile -> Vs (overlaps S gemm + softmax)
    async_copy_128B(ldsV, (unsigned)((lrow * HW + n0 + lcol) * 2), Vg);

    // S = Q * K^T : this wave's 16 m-rows x 128 n-cols
    v8f s[8];
#pragma unroll
    for (int i = 0; i < 8; ++i) s[i] = vzero8();
#pragma unroll
    for (int k0 = 0; k0 < HD; k0 += 32) {
      v16h a = frag_a(Qs + (wv * 16) * FPAD + k0, FPAD);
#pragma unroll
      for (int nt = 0; nt < 8; ++nt) {
        v16h bf = frag_b(Ks + (nt * 16) * FPAD + k0, FPAD);
        s[nt] = wmma16(a, bf, s[nt]);
      }
    }
    // dump scaled S fragments into this wave's LDS tile (f16)
#pragma unroll
    for (int nt = 0; nt < 8; ++nt)
#pragma unroll
      for (int r = 0; r < 8; ++r)
        Ssh[wv][(r + ro) * FPAD + nt * 16 + nl] = (_Float16)(s[nt][r] * scale);
    wait_ds0();

    // online softmax: both half-waves; lane handles row (lane&15), 64 cols
    {
      const int r = lane & 15;
      const int half = lane >> 4;
      _Float16* row = &Ssh[wv][r * FPAD + half * 64];
      v8h buf[8];
#pragma unroll
      for (int j = 0; j < 8; ++j) buf[j] = ((const v8h*)row)[j];
      float pm = -1e30f;
#pragma unroll
      for (int j = 0; j < 8; ++j)
#pragma unroll
        for (int e = 0; e < 8; ++e) pm = fmaxf(pm, (float)buf[j][e]);
      pmax[wv][lane] = pm;
      wait_ds0();
      const float oldm = rowM[wv][r];
      const float mx = fmaxf(oldm, fmaxf(pmax[wv][r], pmax[wv][r + 16]));
      float sum = 0.0f;
#pragma unroll
      for (int j = 0; j < 8; ++j) {
        v8h pv;
#pragma unroll
        for (int e = 0; e < 8; ++e) {
          float ex = __expf((float)buf[j][e] - mx);
          sum += ex;
          pv[e] = (_Float16)ex;
        }
        ((v8h*)row)[j] = pv;
      }
      psum[wv][lane] = sum;
      wait_ds0();
      if (lane < 16) {
        const float corr = __expf(oldm - mx);
        rowC[wv][r] = corr;
        rowL[wv][r] = rowL[wv][r] * corr + psum[wv][r] + psum[wv][r + 16];
        rowM[wv][r] = mx;
      }
      wait_ds0();
    }

    // rescale O accumulators by per-row correction
    float cr[8];
#pragma unroll
    for (int r = 0; r < 8; ++r) cr[r] = rowC[wv][r + ro];
#pragma unroll
    for (int dt = 0; dt < 8; ++dt)
#pragma unroll
      for (int r = 0; r < 8; ++r) o[dt][r] *= cr[r];

    wait_async0();
    __syncthreads();  // V tile visible; all waves done reading Ks

    // prefetch next K tile -> Ks (overlaps O gemm)
    if (n0 + 128 < HW)
      async_copy_128B(ldsK, (unsigned)(((n0 + 128 + lrow) * HD + lcol) * 2), Kg);

    // O += P * V^T : A = P (16m x 128n), B column d contiguous over n
#pragma unroll
    for (int k0 = 0; k0 < 128; k0 += 32) {
      v16h a = frag_a(&Ssh[wv][0] + k0, FPAD);
#pragma unroll
      for (int dt = 0; dt < 8; ++dt) {
        v16h bf = frag_b(Vs + (dt * 16) * FPAD + k0, FPAD);
        o[dt] = wmma16(a, bf, o[dt]);
      }
    }
  }

  // epilogue: O / l, store to aoT[b][m][head*128 + d]
  if (lane < 16) rowC[wv][lane] = 1.0f / rowL[wv][lane];
  wait_ds0();

  const int b = bh >> 2;
  const int head = bh & 3;
  _Float16* dst = aoT + (size_t)b * HW * CH + head * HD;
  float li[8];
#pragma unroll
  for (int r = 0; r < 8; ++r) li[r] = rowC[wv][r + ro];
#pragma unroll
  for (int dt = 0; dt < 8; ++dt)
#pragma unroll
    for (int r = 0; r < 8; ++r) {
      const int m = m0 + wv * 16 + r + ro;
      const int d = dt * 16 + nl;
      dst[(size_t)m * CH + d] = (_Float16)(o[dt][r] * li[r]);
    }
}

// ---------------------------------------------------------------------------
// Kernel 4: proj GEMM + bias + residual -> out fp32 [b][c][hw]
// ---------------------------------------------------------------------------
__global__ __launch_bounds__(256) void k_proj(const _Float16* __restrict__ W,
                                              const _Float16* __restrict__ X,
                                              const float* __restrict__ bias,
                                              const float* __restrict__ xres,
                                              float* __restrict__ out) {
  __shared__ _Float16 As[128 * LDT];
  __shared__ _Float16 Bs[128 * LDT];
  const int b = blockIdx.z;
  const int M0 = blockIdx.x * 128;
  const int N0 = blockIdx.y * 128;
  const int wv = threadIdx.x >> 5;
  const int lane = threadIdx.x & 31;
  const _Float16* Xb = X + (size_t)b * HW * CH;

  v8f acc[8];
#pragma unroll
  for (int i = 0; i < 8; ++i) acc[i] = vzero8();

  const int lrow = threadIdx.x >> 1;
  const int lcol = (threadIdx.x & 1) * 16;

  for (int k0 = 0; k0 < CH; k0 += 32) {
    __syncthreads();
    {
      const v8h* srcA = (const v8h*)(W + (size_t)(M0 + lrow) * CH + k0 + lcol);
      v8h* dstA = (v8h*)(As + lrow * LDT + lcol);
      dstA[0] = srcA[0];
      dstA[1] = srcA[1];
      const v8h* srcB = (const v8h*)(Xb + (size_t)(N0 + lrow) * CH + k0 + lcol);
      v8h* dstB = (v8h*)(Bs + lrow * LDT + lcol);
      dstB[0] = srcB[0];
      dstB[1] = srcB[1];
    }
    __syncthreads();
    v16h a = frag_a(As + (wv * 16) * LDT, LDT);
#pragma unroll
    for (int nt = 0; nt < 8; ++nt) {
      v16h bf = frag_b(Bs + (nt * 16) * LDT, LDT);
      acc[nt] = wmma16(a, bf, acc[nt]);
    }
  }

  const int nl = lane & 15;
  const int ro = (lane >> 4) * 8;
#pragma unroll
  for (int nt = 0; nt < 8; ++nt) {
    const int hw = N0 + nt * 16 + nl;
#pragma unroll
    for (int r = 0; r < 8; ++r) {
      const int o2 = M0 + wv * 16 + r + ro;
      const size_t idx = ((size_t)b * CH + o2) * HW + hw;
      out[idx] = acc[nt][r] + bias[o2] + xres[idx];
    }
  }
}

// ---------------------------------------------------------------------------
// Host launcher
// ---------------------------------------------------------------------------
extern "C" void kernel_launch(void* const* d_in, const int* in_sizes, int n_in,
                              void* d_out, int out_size, void* d_ws, size_t ws_size,
                              hipStream_t stream) {
  const float* x      = (const float*)d_in[0];
  const float* gn_w   = (const float*)d_in[1];
  const float* gn_b   = (const float*)d_in[2];
  const float* qkv_w  = (const float*)d_in[3];
  const float* qkv_b  = (const float*)d_in[4];
  const float* proj_w = (const float*)d_in[5];
  const float* proj_b = (const float*)d_in[6];
  float* out = (float*)d_out;

  const size_t SZ_XNT = (size_t)16 * HW * CH * 2;
  const size_t SZ_WQ  = (size_t)3 * CH * CH * 2;
  const size_t SZ_WP  = (size_t)CH * CH * 2;
  const size_t SZ_QKV = (size_t)16 * NH * HW * HD * 2;
  char* ws = (char*)d_ws;
  size_t off = 0;
  _Float16* xnT  = (_Float16*)(ws + off); off += SZ_XNT;
  _Float16* wq16 = (_Float16*)(ws + off); off += SZ_WQ;
  _Float16* wp16 = (_Float16*)(ws + off); off += SZ_WP;
  _Float16* qT   = (_Float16*)(ws + off); off += SZ_QKV;
  _Float16* kT   = (_Float16*)(ws + off); off += SZ_QKV;
  _Float16* vB   = (_Float16*)(ws + off); off += SZ_QKV;
  _Float16* aoT  = (_Float16*)(ws + off); off += SZ_XNT;
  if (ws_size < off) return;

  k_f32tof16<<<(3 * CH * CH + 255) / 256, 256, 0, stream>>>(qkv_w, wq16, 3 * CH * CH);
  k_f32tof16<<<(CH * CH + 255) / 256, 256, 0, stream>>>(proj_w, wp16, CH * CH);
  k_groupnorm<<<16 * 32, 256, 0, stream>>>(x, gn_w, gn_b, xnT);
  k_gemm_qkv<<<dim3(12, 8, 16), 256, 0, stream>>>(wq16, xnT, qkv_b, qT, kT, vB);
  k_attn<<<dim3(8, 64), 256, 0, stream>>>(qT, kT, vB, aoT);
  k_proj<<<dim3(4, 8, 16), 256, 0, stream>>>(wp16, aoT, proj_b, x, out);
}
